// GoSpecificWattiPooling_47278999994497
// MI455X (gfx1250) — matrix operand order repaired and checked by reading, not compiled, not run on hardware
//
#include <hip/hip_runtime.h>

typedef __bf16 bf16;
typedef __attribute__((ext_vector_type(16))) __bf16 v16bf;
typedef __attribute__((ext_vector_type(8)))  __bf16 v8bf;
typedef __attribute__((ext_vector_type(8)))  float  v8f;

#define NB 8
#define LL 4096
#define TT 512
#define DH 1280
#define DG 768
#define PP 256
#define SCALE 0.0625f   // 256^-0.5

// ---------- fragment helpers (layouts per CDNA5 ISA §7.12.2) ----------

// A fragment 16x32 bf16 from row-major bf16 [M][ld] at (m0, k0)
__device__ inline v16bf load_a_bf16(const bf16* __restrict__ base, int ld, int m0, int k0, int lane) {
    int m = lane & 15;
    int koff = (lane < 16) ? 0 : 8;
    const bf16* p = base + (size_t)(m0 + m) * ld + k0 + koff;
    v8bf lo = *(const v8bf*)(p);        // K = k0+koff .. +7
    v8bf hi = *(const v8bf*)(p + 16);   // K = k0+16+koff .. +7
    v16bf a;
#pragma unroll
    for (int i = 0; i < 8; ++i) { a[i] = lo[i]; a[i + 8] = hi[i]; }
    return a;
}

// A fragment 16x32 from row-major f32 [M][ld], converted to bf16 in-register
__device__ inline v16bf load_a_f32(const float* __restrict__ base, int ld, int m0, int k0, int lane) {
    int m = lane & 15;
    int koff = (lane < 16) ? 0 : 8;
    const float* p = base + (size_t)(m0 + m) * ld + k0 + koff;
    float4 x0 = *(const float4*)(p);
    float4 x1 = *(const float4*)(p + 4);
    float4 x2 = *(const float4*)(p + 16);
    float4 x3 = *(const float4*)(p + 20);
    v16bf a;
    a[0]  = (bf16)x0.x; a[1]  = (bf16)x0.y; a[2]  = (bf16)x0.z; a[3]  = (bf16)x0.w;
    a[4]  = (bf16)x1.x; a[5]  = (bf16)x1.y; a[6]  = (bf16)x1.z; a[7]  = (bf16)x1.w;
    a[8]  = (bf16)x2.x; a[9]  = (bf16)x2.y; a[10] = (bf16)x2.z; a[11] = (bf16)x2.w;
    a[12] = (bf16)x3.x; a[13] = (bf16)x3.y; a[14] = (bf16)x3.z; a[15] = (bf16)x3.w;
    return a;
}

// B fragment 32(K)x16(N): element (k,n) = src[n0+n][k0+k], src row-major [N][ld]
__device__ inline v16bf load_b_bf16(const bf16* __restrict__ base, int ld, int n0, int k0, int lane) {
    int n = lane & 15;
    int koff = (lane < 16) ? 0 : 8;
    const bf16* p = base + (size_t)(n0 + n) * ld + k0 + koff;
    v8bf lo = *(const v8bf*)(p);
    v8bf hi = *(const v8bf*)(p + 16);
    v16bf b;
#pragma unroll
    for (int i = 0; i < 8; ++i) { b[i] = lo[i]; b[i + 8] = hi[i]; }
    return b;
}

__device__ inline v8f wmma_bf16(v16bf a, v16bf b, v8f c) {
    return __builtin_amdgcn_wmma_f32_16x16x32_bf16(false, a, false, b, (short)0, c, false, false);
}

// ---------- kernel 0: f32 -> bf16 weight convert ----------
__global__ void cvt_f32_bf16(const float* __restrict__ x, bf16* __restrict__ y, int n) {
    int i = blockIdx.x * blockDim.x + threadIdx.x;
    if (i < n) y[i] = (bf16)x[i];
}

// ---------- kernel 1: Y[b][m][p] = scale * sum_d X[b][m][d] * W[p][d]  (bf16 out) ----------
__global__ void __launch_bounds__(256) proj_kernel(const float* __restrict__ X, const bf16* __restrict__ W,
                                                   bf16* __restrict__ Y, int M, int Kd, float scale, int totalTiles) {
    int lane = threadIdx.x & 31;
    int w = blockIdx.x * (blockDim.x >> 5) + (threadIdx.x >> 5);
    if (w >= totalTiles) return;              // wave-uniform exit, EXEC stays full for WMMA
    int nt = w & 3;                           // 4 n-tiles of 64 (PP = 256)
    int rest = w >> 2;
    int mtPerB = M >> 4;
    int mt = rest % mtPerB;
    int b  = rest / mtPerB;
    const float* Xb = X + (size_t)b * M * Kd;
    v8f c[4] = {};
    for (int k = 0; k < Kd; k += 32) {
        v16bf a = load_a_f32(Xb, Kd, mt * 16, k, lane);
#pragma unroll
        for (int j = 0; j < 4; ++j) {
            v16bf bm = load_b_bf16(W, Kd, nt * 64 + j * 16, k, lane);
            c[j] = wmma_bf16(a, bm, c[j]);
        }
    }
    bf16* Yb = Y + (size_t)b * M * PP;
    int col0 = nt * 64 + (lane & 15);
    int row0 = mt * 16 + ((lane < 16) ? 0 : 8);
#pragma unroll
    for (int j = 0; j < 4; ++j)
#pragma unroll
        for (int r = 0; r < 8; ++r)
            Yb[(size_t)(row0 + r) * PP + col0 + j * 16] = (bf16)(c[j][r] * scale);
}

// ---------- kernel 2: streaming softmax over L, write bf16 probs ----------
__global__ void __launch_bounds__(128) scores_kernel(const bf16* __restrict__ Q, const bf16* __restrict__ K,
                                                     const unsigned char* __restrict__ mask,
                                                     bf16* __restrict__ probs) {
    int lane = threadIdx.x & 31;
    int w = blockIdx.x * (blockDim.x >> 5) + (threadIdx.x >> 5);
    const int mtPerB = TT / 16;
    if (w >= NB * mtPerB) return;
    int mt = w % mtPerB;
    int b  = w / mtPerB;
    const bf16* Qb = Q + (size_t)b * TT * PP;
    const bf16* Kb = K + (size_t)b * LL * PP;
    const float NEG_INF = -__builtin_inff();

    v16bf qf[8];                               // Q rows resident: 16 x 256 bf16
#pragma unroll
    for (int i = 0; i < 8; ++i) qf[i] = load_a_bf16(Qb, PP, mt * 16, i * 32, lane);

    float m8[8], s8[8];
#pragma unroll
    for (int r = 0; r < 8; ++r) { m8[r] = NEG_INF; s8[r] = 0.f; }
    int col = lane & 15;

    // pass A: streaming (max, sumexp) per lane-column, per C-register row
    for (int lt = 0; lt < LL; lt += 16) {
        v8f c = {};
#pragma unroll
        for (int i = 0; i < 8; ++i)
            c = wmma_bf16(qf[i], load_b_bf16(Kb, PP, lt, i * 32, lane), c);
        bool mk = mask[(size_t)b * LL + lt + col] != 0;
#pragma unroll
        for (int r = 0; r < 8; ++r) {
            float v  = mk ? NEG_INF : c[r];
            float mn = fmaxf(m8[r], v);
            if (mn != NEG_INF) {
                s8[r] = s8[r] * __expf(m8[r] - mn) + __expf(v - mn);
                m8[r] = mn;
            }
        }
    }
    // merge (m,s) across the 16 lanes sharing each row (xor<16 stays within halves)
#pragma unroll
    for (int off = 1; off < 16; off <<= 1) {
#pragma unroll
        for (int r = 0; r < 8; ++r) {
            float mo = __shfl_xor(m8[r], off, 32);
            float so = __shfl_xor(s8[r], off, 32);
            float mn = fmaxf(m8[r], mo);
            float s  = (mn == NEG_INF) ? 0.f
                     : s8[r] * __expf(m8[r] - mn) + so * __expf(mo - mn);
            m8[r] = mn; s8[r] = s;
        }
    }
    float rinv[8];
#pragma unroll
    for (int r = 0; r < 8; ++r) rinv[r] = (s8[r] > 0.f) ? 1.f / s8[r] : 0.f;

    // pass B: recompute S, emit probabilities
    bf16* Pb = probs + (size_t)b * TT * LL;
    int row0 = mt * 16 + ((lane < 16) ? 0 : 8);
    for (int lt = 0; lt < LL; lt += 16) {
        v8f c = {};
#pragma unroll
        for (int i = 0; i < 8; ++i)
            c = wmma_bf16(qf[i], load_b_bf16(Kb, PP, lt, i * 32, lane), c);
        bool mk = mask[(size_t)b * LL + lt + col] != 0;
#pragma unroll
        for (int r = 0; r < 8; ++r) {
            float pv = mk ? 0.f : __expf(c[r] - m8[r]) * rinv[r];
            Pb[(size_t)(row0 + r) * LL + lt + col] = (bf16)pv;
        }
    }
}

// ---------- kernel 3: Z = probs @ H  (H staged transposed to bf16 in LDS) ----------
__global__ void __launch_bounds__(256) z_kernel(const bf16* __restrict__ probs, const float* __restrict__ H,
                                                float* __restrict__ Z) {
    __shared__ bf16 hT[128 * 40];             // [d_local][l_local], stride 40 (16B-aligned halves)
    int b  = blockIdx.z;
    int d0 = blockIdx.y * 128;
    int m0 = blockIdx.x * 64;
    int lane = threadIdx.x & 31;
    int wave = threadIdx.x >> 5;
    int mw = wave >> 1;                       // 0..3
    int nw = wave & 1;                        // 0..1
    const bf16*  Pb = probs + (size_t)b * TT * LL;
    const float* Hb = H + (size_t)b * LL * DH;

    int l_local = threadIdx.x >> 3;           // 0..31
    int dg = (threadIdx.x & 7) * 16;          // 0..112
    v8f c[4] = {};

    for (int l0 = 0; l0 < LL; l0 += 32) {
        const float* hrow = Hb + (size_t)(l0 + l_local) * DH + d0 + dg;
#pragma unroll
        for (int j = 0; j < 4; ++j) {
            float4 v = *(const float4*)(hrow + j * 4);
            int dd = dg + j * 4;
            hT[(dd + 0) * 40 + l_local] = (bf16)v.x;
            hT[(dd + 1) * 40 + l_local] = (bf16)v.y;
            hT[(dd + 2) * 40 + l_local] = (bf16)v.z;
            hT[(dd + 3) * 40 + l_local] = (bf16)v.w;
        }
        if (l0 + 32 < LL)
            __builtin_prefetch(Hb + (size_t)(l0 + 32 + l_local) * DH + d0 + dg, 0, 1);
        __syncthreads();

        v16bf a = load_a_bf16(Pb, LL, m0 + mw * 16, l0, lane);
#pragma unroll
        for (int j = 0; j < 4; ++j) {
            int nloc = nw * 64 + j * 16 + (lane & 15);
            const bf16* p = &hT[nloc * 40 + ((lane < 16) ? 0 : 8)];
            v8bf lo = *(const v8bf*)(p);
            v8bf hi = *(const v8bf*)(p + 16);
            v16bf bm;
#pragma unroll
            for (int i = 0; i < 8; ++i) { bm[i] = lo[i]; bm[i + 8] = hi[i]; }
            c[j] = wmma_bf16(a, bm, c[j]);
        }
        __syncthreads();
    }

    int row0 = m0 + mw * 16 + ((lane < 16) ? 0 : 8);
#pragma unroll
    for (int j = 0; j < 4; ++j) {
        int d = d0 + nw * 64 + j * 16 + (lane & 15);
#pragma unroll
        for (int r = 0; r < 8; ++r)
            Z[((size_t)b * TT + row0 + r) * DH + d] = c[j][r];
    }
}

// ---------- launcher ----------
extern "C" void kernel_launch(void* const* d_in, const int* in_sizes, int n_in,
                              void* d_out, int out_size, void* d_ws, size_t ws_size,
                              hipStream_t stream) {
    const float*         Hin  = (const float*)d_in[0];
    const float*         Gin  = (const float*)d_in[1];
    const unsigned char* mask = (const unsigned char*)d_in[2];
    const float*         Wk   = (const float*)d_in[3];
    const float*         Wq   = (const float*)d_in[4];
    float*               Z    = (float*)d_out;

    char* ws = (char*)d_ws;
    size_t off = 0;
    auto take = [&](size_t bytes) { char* p = ws + off; off += (bytes + 255) & ~(size_t)255; return p; };
    bf16* Wk_bf = (bf16*)take((size_t)PP * DH * 2);
    bf16* Wq_bf = (bf16*)take((size_t)PP * DG * 2);
    bf16* Qb    = (bf16*)take((size_t)NB * TT * PP * 2);
    bf16* Kb    = (bf16*)take((size_t)NB * LL * PP * 2);
    bf16* Pr    = (bf16*)take((size_t)NB * TT * LL * 2);   // ~53.5 MB total
    (void)ws_size; (void)in_sizes; (void)n_in; (void)out_size;

    { int n = PP * DH; cvt_f32_bf16<<<(n + 255) / 256, 256, 0, stream>>>(Wk, Wk_bf, n); }
    { int n = PP * DG; cvt_f32_bf16<<<(n + 255) / 256, 256, 0, stream>>>(Wq, Wq_bf, n); }
    { int tiles = NB * (LL / 16) * 4;
      proj_kernel<<<(tiles + 7) / 8, 256, 0, stream>>>(Hin, Wk_bf, Kb, LL, DH, 1.0f, tiles); }
    { int tiles = NB * (TT / 16) * 4;
      proj_kernel<<<(tiles + 7) / 8, 256, 0, stream>>>(Gin, Wq_bf, Qb, TT, DG, SCALE, tiles); }
    { int wt = NB * (TT / 16);
      scores_kernel<<<wt / 4, 128, 0, stream>>>(Qb, Kb, mask, Pr); }
    { dim3 g(TT / 64, DH / 128, NB);
      z_kernel<<<g, 256, 0, stream>>>(Pr, Hin, Z); }
}
